// TreeMLPV3_5265629905476
// MI455X (gfx1250) — compile-verified, matrix operand order checked by reading, not run on hardware
//
#include <hip/hip_runtime.h>

#define N_EMBD 384
#define D_FF   1536
#define RCH    64
#define NLVL   6
#define BB     16
#define HH     64
#define WW     64
#define NTOK   (BB*HH*WW)   // 65536

typedef __bf16 bf16;
typedef bf16  bf16x16 __attribute__((ext_vector_type(16)));
typedef float floatx8 __attribute__((ext_vector_type(8)));
typedef unsigned int u32x4 __attribute__((ext_vector_type(4)));
typedef int i32x4 __attribute__((ext_vector_type(4)));
typedef int i32x8 __attribute__((ext_vector_type(8)));

#if defined(__has_builtin)
#  if __has_builtin(__builtin_amdgcn_tensor_load_to_lds)
#    define HAVE_TDM 1
#  else
#    define HAVE_TDM 0
#  endif
#else
#  define HAVE_TDM 0
#endif

union FragB16 { bf16x16 v; uint4 q[2]; };

__device__ __forceinline__ unsigned short f2bf(float f) {
  union { float f; unsigned u; } c; c.f = f;
  unsigned r = (c.u + 0x7FFFu + ((c.u >> 16) & 1u)) >> 16;   // RNE
  return (unsigned short)r;
}

__device__ __forceinline__ float gelu_tanh(float v) {
  const float k = 0.7978845608028654f;  // sqrt(2/pi)
  return 0.5f * v * (1.0f + tanhf(k * (v + 0.044715f * v * v * v)));
}

// ---- CDNA5 async global->LDS (ASYNCcnt); per-lane 16B, bypasses VGPRs ------
__device__ __forceinline__ void async_ld16(void* lds_dst, const void* gsrc) {
  asm volatile("global_load_async_to_lds_b128 %0, %1, off"
               :: "v"((unsigned)(unsigned long long)lds_dst),
                  "v"((unsigned long long)gsrc)
               : "memory");
}
__device__ __forceinline__ void wait_async0() {
  asm volatile("s_wait_asynccnt 0x0" ::: "memory");
}

// ---- CDNA5 TDM: 2D tile (64 rows x 128 bf16) of w2T -> LDS with padding ----
// LDS row stride 136 elements = 64 payload DWORDs + 4 pad DWORDs per row.
__device__ __forceinline__ void tdm_load_w2_chunk(const unsigned short* gsrc,
                                                  unsigned lds_addr)
{
#if HAVE_TDM
  const unsigned long long ga = (unsigned long long)gsrc;
  u32x4 g0;
  g0.x = 1u;                                                  // count=1 user D#
  g0.y = lds_addr;                                            // LDS byte address
  g0.z = (unsigned)ga;                                        // global_addr[31:0]
  g0.w = (unsigned)((ga >> 32) & 0x01FFFFFFull) | (2u << 30); // addr[56:32], type=2
  i32x8 g1;
  g1[0] = (1 << 16)      // data_size = 2 bytes
        | (1 << 20)      // pad_enable
        | (5 << 22)      // pad_interval: 64 DWORDs
        | (3 << 25);     // pad_amount:   4 DWORDs
  g1[1] = (int)(1536u << 16);   // tensor_dim0[15:0]=1536  (atomic_barrier_addr=0)
  g1[2] = (int)(64u << 16);     // tensor_dim0 hi=0 ; tensor_dim1[15:0]=64
  g1[3] = (int)(128u << 16);    // tensor_dim1 hi=0 ; tile_dim0=128
  g1[4] = 64;                   // tile_dim1=64 ; tile_dim2=0
  g1[5] = 1536;                 // tensor_dim0_stride[31:0]
  g1[6] = 0;                    // stride hi / tensor_dim1_stride lo
  g1[7] = 0;
  const i32x4 z4 = {0, 0, 0, 0};
#if __clang_major__ >= 23
  const i32x8 z8 = {0, 0, 0, 0, 0, 0, 0, 0};
  __builtin_amdgcn_tensor_load_to_lds(g0, g1, z4, z4, z8, 0);
#else
  __builtin_amdgcn_tensor_load_to_lds(g0, g1, z4, z4, 0);
#endif
#else
  (void)gsrc; (void)lds_addr;
#endif
}

// ---------------- prep: weights -> bf16, transposed to K-major rows ----------
__global__ void prep_weights(const float* __restrict__ w1, const float* __restrict__ w2,
                             unsigned short* __restrict__ w1T, unsigned short* __restrict__ w2T)
{
  const int i  = blockIdx.x * 256 + threadIdx.x;
  const int N1 = N_EMBD * D_FF;                 // 589824
  if (i < N1) {
    const int n = i / N_EMBD, k = i % N_EMBD;   // w1T[n][k] = w1[k][n]
    w1T[i] = f2bf(w1[k * D_FF + n]);
  } else {
    const int e = i - N1;
    if (e < RCH * D_FF) {
      const int n = e / D_FF, k = e % D_FF;     // w2T[n][k] = w2[k][n]
      w2T[e] = f2bf(w2[k * RCH + n]);
    }
  }
}

// ---------------- prep: x fp32 -> bf16 --------------------------------------
__global__ void cvt_x(const float* __restrict__ x, unsigned short* __restrict__ xb, int n4)
{
  const int i = blockIdx.x * 256 + threadIdx.x;
  if (i < n4) {
    const float4 v = ((const float4*)x)[i];
    ushort4 o;
    o.x = f2bf(v.x); o.y = f2bf(v.y); o.z = f2bf(v.z); o.w = f2bf(v.w);
    ((ushort4*)xb)[i] = o;
  }
}

// ---------------- prep: sibling-average pyramid, order i in 1..5 ------------
// layout: [b][h>>i][w>>i][i*64 channels], bf16
__global__ void avg_k(const float* __restrict__ x, unsigned short* __restrict__ out,
                      int order, int nOut)
{
  const int oid = blockIdx.x * 256 + threadIdx.x;
  if (oid >= nOut) return;
  const int C  = order * RCH;
  const int Wb = WW >> order, Hb = HH >> order, bs = 1 << order;
  const int c  = oid % C;
  const int wb = (oid / C) % Wb;
  const int hb = (oid / (C * Wb)) % Hb;
  const int b  =  oid / (C * Wb * Hb);
  float s = 0.f;
  for (int dh = 0; dh < bs; ++dh)
    for (int dw = 0; dw < bs; ++dw)
      s += x[((size_t)((b * HH + hb * bs + dh) * WW) + wb * bs + dw) * N_EMBD + c];
  out[oid] = f2bf(s * (1.0f / (float)(bs * bs)));
}

// ---------------- fused tree-MLP: one WG = 128 tokens × one level -----------
__global__ void __launch_bounds__(256, 1)
tree_mlp_fused(const unsigned short* __restrict__ xb,
               const unsigned short* __restrict__ w1T,
               const unsigned short* __restrict__ w2T,
               const float* __restrict__ b1,
               const float* __restrict__ b2,
               const unsigned short* __restrict__ a1,
               const unsigned short* __restrict__ a2,
               const unsigned short* __restrict__ a3,
               const unsigned short* __restrict__ a4,
               const unsigned short* __restrict__ a5,
               float* __restrict__ out)
{
  constexpr int M  = 128;
  constexpr int SA = 392;   // merged LDS row stride (16B-aligned rows)
  constexpr int SH = 136;   // hid-chunk LDS stride
  constexpr int SW = 136;   // staged-w2 LDS stride (matches TDM padding)
  extern __shared__ unsigned short smem[];
  unsigned short* smA = smem;                 // [128][392] merged tile, bf16
  unsigned short* smH = smem + M * SA;        // [128][136] gelu'd hid chunk, bf16
  unsigned short* smW = smem + M * SA + M*SH; // [64][136]  staged w2 chunk, bf16

  const int lvl  = blockIdx.y;
  const int t0   = blockIdx.x * M;
  const int tid  = threadIdx.x;
  const int lane = tid & 31;
  const int wave = tid >> 5;     // 8 waves
  const int n16  = lane & 15;
  const int hi   = lane >> 4;

  const unsigned short* ab =
      (lvl == 1) ? a1 : (lvl == 2) ? a2 : (lvl == 3) ? a3 : (lvl == 4) ? a4 : a5;

  // ---- stage merged tile (128 tokens × 384 feats) into LDS, async ----------
  {
    const int row  = tid >> 1;          // token row 0..127
    const int half = tid & 1;           // channel group (3 ch each)
    const int t = t0 + row;
    const int b = t >> 12, h = (t >> 6) & 63, w = t & 63;
    const int Hb = HH >> lvl, Wb = WW >> lvl;
    const size_t blk = lvl ? ((size_t)((b * Hb + (h >> lvl)) * Wb + (w >> lvl)) * (lvl * RCH))
                           : (size_t)0;
    #pragma unroll
    for (int jc = 0; jc < 3; ++jc) {
      const int j = half * 3 + jc;
      const unsigned short* src = (j >= lvl)
          ? (xb + (size_t)t * N_EMBD + j * RCH)   // raw channel
          : (ab + blk + j * RCH);                 // block-averaged channel
      unsigned short* dst = smA + row * SA + j * RCH;
      #pragma unroll
      for (int q = 0; q < 8; ++q)
        async_ld16(dst + q * 8, src + q * 8);     // 16B/lane direct to LDS
    }
  }

  const floatx8 zero = {0.f,0.f,0.f,0.f,0.f,0.f,0.f,0.f};
  floatx8 oacc[4];
  #pragma unroll
  for (int i = 0; i < 4; ++i) oacc[i] = zero;

  wait_async0();          // merged tile landed
  __syncthreads();

  for (int nc = 0; nc < D_FF / 128; ++nc) {
    const int C0 = nc * 128;

    if (nc) __syncthreads();   // previous chunk's smH/smW readers are done

    // kick the w2 chunk DMA; it overlaps with all of GEMM1 below
#if HAVE_TDM
    if (wave == 0)
      tdm_load_w2_chunk(w2T + C0, (unsigned)(unsigned long long)(void*)smW);
#else
    #pragma unroll
    for (int p = 0; p < 4; ++p) {
      const int idx = tid * 4 + p;   // 0..1023 x 16B
      const int nr  = idx >> 4;      // out col 0..63
      const int qc  = idx & 15;      // 16B step along k
      async_ld16(smW + nr * SW + qc * 8,
                 w2T + (size_t)nr * D_FF + C0 + qc * 8);
    }
#endif

    // ---- GEMM1: 128×128 hid chunk; wave owns cols [wave*16, +16) -----------
    floatx8 acc[8];
    #pragma unroll
    for (int i = 0; i < 8; ++i) acc[i] = zero;

    const unsigned short* bRow = w1T + (size_t)(C0 + wave * 16 + n16) * N_EMBD;

    for (int kk = 0; kk < N_EMBD; kk += 32) {
      __builtin_prefetch(bRow + kk + 64, 0, 1);       // global_prefetch next K
      FragB16 fb;
      fb.q[0] = *(const uint4*)(bRow + kk + hi * 8);
      fb.q[1] = *(const uint4*)(bRow + kk + 16 + hi * 8);
      #pragma unroll
      for (int rt = 0; rt < 8; ++rt) {
        FragB16 fa;
        const unsigned short* aRow = smA + (rt * 16 + n16) * SA + kk;
        fa.q[0] = *(const uint4*)(aRow + hi * 8);
        fa.q[1] = *(const uint4*)(aRow + 16 + hi * 8);
        acc[rt] = __builtin_amdgcn_wmma_f32_16x16x32_bf16(
            false, fa.v, false, fb.v, (short)0, acc[rt], false, false);
      }
    }

    // ---- epilogue: +b1, gelu, -> bf16 smH ----------------------------------
    const float b1v = b1[C0 + wave * 16 + n16];       // lane's fixed column
    #pragma unroll
    for (int rt = 0; rt < 8; ++rt) {
      #pragma unroll
      for (int r = 0; r < 8; ++r) {
        const int row = rt * 16 + hi * 8 + r;
        smH[row * SH + wave * 16 + n16] = f2bf(gelu_tanh(acc[rt][r] + b1v));
      }
    }

    // DMA / async staging must have landed before anyone reads smW
#if HAVE_TDM
    if (wave == 0) __builtin_amdgcn_s_wait_tensorcnt(0);
#else
    wait_async0();
#endif
    __syncthreads();   // smH + smW visible

    // ---- GEMM2 partial: out[128×64] += hid_chunk @ w2_chunk ----------------
    for (int k2 = 0; k2 < 128; k2 += 32) {
      FragB16 fa;
      const unsigned short* aRow = smH + (wave * 16 + n16) * SH + k2;
      fa.q[0] = *(const uint4*)(aRow + hi * 8);
      fa.q[1] = *(const uint4*)(aRow + 16 + hi * 8);
      #pragma unroll
      for (int ct = 0; ct < 4; ++ct) {
        FragB16 fb;
        const unsigned short* wRow = smW + (ct * 16 + n16) * SW + k2;
        fb.q[0] = *(const uint4*)(wRow + hi * 8);
        fb.q[1] = *(const uint4*)(wRow + 16 + hi * 8);
        oacc[ct] = __builtin_amdgcn_wmma_f32_16x16x32_bf16(
            false, fa.v, false, fb.v, (short)0, oacc[ct], false, false);
      }
    }
  }

  // ---- write out: wave rows [wave*16,+16), cols 0..63, +b2 -----------------
  #pragma unroll
  for (int ct = 0; ct < 4; ++ct) {
    const int col  = ct * 16 + n16;
    const float bv = b2[col];
    #pragma unroll
    for (int r = 0; r < 8; ++r) {
      const int row = wave * 16 + hi * 8 + r;
      const int t   = t0 + row;
      out[((size_t)t * NLVL + lvl) * RCH + col] = oacc[ct][r] + bv;
    }
  }
}

// ---------------- launcher ---------------------------------------------------
extern "C" void kernel_launch(void* const* d_in, const int* in_sizes, int n_in,
                              void* d_out, int out_size, void* d_ws, size_t ws_size,
                              hipStream_t stream)
{
  const float* x  = (const float*)d_in[0];
  const float* w1 = (const float*)d_in[1];
  const float* b1 = (const float*)d_in[2];
  const float* w2 = (const float*)d_in[3];
  const float* b2 = (const float*)d_in[4];
  float* out = (float*)d_out;
  char*  ws  = (char*)d_ws;

  // workspace layout (bytes), all 16B aligned; total ≈ 55.4 MB
  unsigned short* w1T = (unsigned short*)(ws + 0);          // 1,179,648 B
  unsigned short* w2T = (unsigned short*)(ws + 1179648);    //   196,608 B
  unsigned short* xb  = (unsigned short*)(ws + 1376256);    // 50,331,648 B
  unsigned short* a1  = (unsigned short*)(ws + 51707904);   // 2,097,152 B
  unsigned short* a2  = (unsigned short*)(ws + 53805056);   // 1,048,576 B
  unsigned short* a3  = (unsigned short*)(ws + 54853632);   //   393,216 B
  unsigned short* a4  = (unsigned short*)(ws + 55246848);   //   131,072 B
  unsigned short* a5  = (unsigned short*)(ws + 55377920);   //    40,960 B

  prep_weights<<<2688, 256, 0, stream>>>(w1, w2, w1T, w2T);
  cvt_x<<<(NTOK * N_EMBD / 4 + 255) / 256, 256, 0, stream>>>(x, xb, NTOK * N_EMBD / 4);

  unsigned short* avgs[5] = {a1, a2, a3, a4, a5};
  for (int i = 1; i <= 5; ++i) {
    const int nOut = BB * (HH >> i) * (WW >> i) * (i * RCH);
    avg_k<<<(nOut + 255) / 256, 256, 0, stream>>>(x, avgs[i - 1], i, nOut);
  }

  const size_t lds = (size_t)(128 * 392 + 128 * 136 + 64 * 136) * sizeof(unsigned short);
  tree_mlp_fused<<<dim3(NTOK / 128, NLVL), 256, lds, stream>>>(
      xb, w1T, w2T, b1, b2, a1, a2, a3, a4, a5, out);
}
